// AttentionMechanism_34540126995074
// MI455X (gfx1250) — matrix operand order repaired and checked by reading, not compiled, not run on hardware
//
#include <hip/hip_runtime.h>

// ---------------------------------------------------------------------------
// Problem constants (from reference): B=32, T=1500, E=D=A=512, C=10, K=201
// ---------------------------------------------------------------------------
#define NB    32
#define NT    1500
#define NE    512
#define NA    512
#define NC    10
#define NK    201
#define ROWS  (NB * NT)        // 48000 rows of the big GEMM
#define KX    544              // K extended: 512 (enc) + 32 (conv feat, padded)
#define KSTEPS 17              // 544 / 32
#define LSTR  552              // LDS row stride in halves (544 + 8 pad, conflict-spread)

typedef __attribute__((ext_vector_type(16))) __bf16       v16bf;
typedef __attribute__((ext_vector_type(8)))  float        v8f;
typedef __attribute__((ext_vector_type(8)))  unsigned int v8u;

__device__ __forceinline__ unsigned short f2bf(float f) {
    unsigned u = __builtin_bit_cast(unsigned, f);
    unsigned rb = ((u >> 16) & 1u) + 0x7FFFu;     // round-to-nearest-even
    return (unsigned short)((u + rb) >> 16);
}

__device__ __forceinline__ v16bf mkv(uint4 lo, uint4 hi) {
    v8u t;
    t[0] = lo.x; t[1] = lo.y; t[2] = lo.z; t[3] = lo.w;
    t[4] = hi.x; t[5] = hi.y; t[6] = hi.z; t[7] = hi.w;
    return __builtin_bit_cast(v16bf, t);
}

// ---------------------------------------------------------------------------
// Kernel 1: pre-swizzle [W_enc ; W_conv ; 0] into WMMA B-operand order (bf16).
// Layout: element index = ((n*17 + s)*32 + lane)*16 + h
//   B operand (16x16x32 bf16): lane L -> column N = n*16 + L%16,
//   halves h=0..15 -> K = 32*s + (h<8 ? 8*hh + h : 16 + 8*hh + h - 8), hh = (L<16 ? 0 : 1)
// ---------------------------------------------------------------------------
__global__ __launch_bounds__(256) void prep_wb(
    const float* __restrict__ Wenc, const float* __restrict__ Wconv,
    unsigned short* __restrict__ Wb)
{
    int idx = blockIdx.x * 256 + threadIdx.x;           // 32*17*32*16 = 278528
    if (idx >= 32 * KSTEPS * 32 * 16) return;
    int h    = idx & 15;
    int lane = (idx >> 4) & 31;
    int q    = idx >> 9;                                // n*17 + s
    int s    = q % KSTEPS;
    int n    = q / KSTEPS;
    int col  = n * 16 + (lane & 15);
    int k    = s * 32 + ((h < 8) ? (((lane < 16) ? 0 : 8) + h) : (16 + ((lane < 16) ? 0 : 8) + h - 8));
    float v = 0.f;
    if (k < 512)            v = Wenc[k * NA + col];
    else if (k - 512 < NC)  v = Wconv[(k - 512) * NA + col];
    Wb[idx] = f2bf(v);
}

// ---------------------------------------------------------------------------
// Kernel 2: dec_row[b,a] = b_enc[a] + sum_d dec_out[b,d] * W_dec[d,a]
// ---------------------------------------------------------------------------
__global__ __launch_bounds__(256) void decrow_k(
    const float* __restrict__ dec, const float* __restrict__ Wdec,
    const float* __restrict__ benc, float* __restrict__ dec_row)
{
    __shared__ float sd[512];
    const int b = blockIdx.y;
    const int a = blockIdx.x * 256 + threadIdx.x;
    for (int i = threadIdx.x; i < 512; i += 256) sd[i] = dec[b * 512 + i];
    __syncthreads();
    float acc = benc[a];
    for (int d = 0; d < 512; ++d) acc = fmaf(sd[d], Wdec[d * NA + a], acc);
    dec_row[b * NA + a] = acc;
}

// ---------------------------------------------------------------------------
// Kernel 3: location conv (K=201, C=10, zero-pad 100), output bf16 padded to 16
// ---------------------------------------------------------------------------
__global__ __launch_bounds__(256) void conv_loc(
    const float* __restrict__ aw, const float* __restrict__ cw,
    unsigned short* __restrict__ cf16)
{
    __shared__ float s_aw[256 + NK - 1];    // 456
    __shared__ float s_w[NC * NK];          // 2010
    const int b = blockIdx.y, t0 = blockIdx.x * 256, tid = threadIdx.x;
    for (int i = tid; i < NC * NK; i += 256) s_w[i] = cw[i];
    for (int i = tid; i < 256 + NK - 1; i += 256) {
        int t = t0 - (NK / 2) + i;
        s_aw[i] = (t >= 0 && t < NT) ? aw[b * NT + t] : 0.f;
    }
    __syncthreads();
    const int t = t0 + tid;
    if (t >= NT) return;
    float acc[NC];
#pragma unroll
    for (int c = 0; c < NC; ++c) acc[c] = 0.f;
    for (int k = 0; k < NK; ++k) {
        float x = s_aw[tid + k];
#pragma unroll
        for (int c = 0; c < NC; ++c) acc[c] = fmaf(x, s_w[c * NK + k], acc[c]);
    }
    alignas(16) unsigned short o[16];
#pragma unroll
    for (int c = 0; c < NC; ++c) o[c] = f2bf(acc[c]);
#pragma unroll
    for (int c = NC; c < 16; ++c) o[c] = 0;
    uint4* dst = reinterpret_cast<uint4*>(cf16 + (size_t)(b * NT + t) * 16);
    dst[0] = *reinterpret_cast<uint4*>(&o[0]);
    dst[1] = *reinterpret_cast<uint4*>(&o[8]);
}

// ---------------------------------------------------------------------------
// Kernel 4: fused energy GEMM, M=32 rows per wave.
//   energy[row] = sum_a V[a] * tanh( (enc@W_enc)[row,a] + dec_row[b,a]
//                                    + (conv_feat@W_conv)[row,a] )
// Conv term folded into WMMA as K-step 16 (K 512..543). Each streamed B
// operand (1 KB) feeds TWO v_wmma on independent accumulator chains, halving
// B-side L2 traffic vs M=16. To stay under 256 VGPRs (so the scheduler keeps
// deep multi-buffered global-load pipelining), only subtile 0's A operands are
// register-resident (17*8 = 136 VGPRs); subtile 1's A stays in LDS and is
// re-read per K-step (2x ds_load_b128). An empty asm on `nbase` stops LICM
// from hoisting those LDS loads back into registers.
// ---------------------------------------------------------------------------
__global__ __launch_bounds__(64) void energy_gemm(
    const float* __restrict__ enc, const unsigned short* __restrict__ Wb,
    const unsigned short* __restrict__ cf16, const float* __restrict__ dec_row,
    const float* __restrict__ Vv, float* __restrict__ energy)
{
    __shared__ unsigned short lds[2 * 16 * LSTR];
    const int lane = threadIdx.x & 31;
    const int wv   = threadIdx.x >> 5;
    const int m0   = (blockIdx.x * 2 + wv) * 32;      // 32 rows per wave
    unsigned short* A = lds + wv * 16 * LSTR;

    const int arow = lane & 15;
    const int aoff = (lane < 16) ? 0 : 8;

    v16bf a0[KSTEPS];
#pragma unroll 1
    for (int half = 0; half < 2; ++half) {
        const int mb = m0 + half * 16;
        // Stage A subtile: 16 rows x 512 enc (f32 -> bf16)
        for (int i = lane; i < 16 * 128; i += 32) {
            int row = i >> 7, c4 = (i & 127) << 2;
            float4 f = *reinterpret_cast<const float4*>(enc + (size_t)(mb + row) * NE + c4);
            uint2 p;
            p.x = (unsigned)f2bf(f.x) | ((unsigned)f2bf(f.y) << 16);
            p.y = (unsigned)f2bf(f.z) | ((unsigned)f2bf(f.w) << 16);
            *reinterpret_cast<uint2*>(&A[row * LSTR + c4]) = p;
        }
        // Stage conv features at K=512..527, zeros at K=528..543
        {
            int row = lane & 15, part = lane >> 4;
            uint4 cfv = *reinterpret_cast<const uint4*>(cf16 + (size_t)(mb + row) * 16 + part * 8);
            *reinterpret_cast<uint4*>(&A[row * LSTR + 512 + part * 8]) = cfv;
            uint4 z; z.x = z.y = z.z = z.w = 0u;
            *reinterpret_cast<uint4*>(&A[row * LSTR + 528 + part * 8]) = z;
        }
        if (half == 0) {
            // Lift subtile 0's 17 A operands to VGPRs (ISA A layout:
            // lane<16 -> K {0..7,16..23}, lane>=16 -> K {8..15,24..31})
#pragma unroll
            for (int s = 0; s < KSTEPS; ++s) {
                uint4 lo = *reinterpret_cast<const uint4*>(&A[arow * LSTR + s * 32 + aoff]);
                uint4 hi = *reinterpret_cast<const uint4*>(&A[arow * LSTR + s * 32 + aoff + 16]);
                a0[s] = mkv(lo, hi);
            }
            // subtile 1 overwrites the buffer next iteration (in-wave LDS
            // ordering guarantees the reads above complete first)
        }
    }

    // C layout: lane<16 -> N=lane, M=r ; lane>=16 -> N=lane-16, M=8+r
    const int rbase = (lane >> 4) << 3;
    const int col0  = lane & 15;
    const int b0    = m0 / NT;
    const int bnd   = (b0 + 1) * NT;          // rows >= bnd belong to batch b0+1
    const int off0  = b0 * NA, off1 = off0 + NA;

    float eb[2][8];
#pragma unroll
    for (int r = 0; r < 8; ++r) { eb[0][r] = 0.f; eb[1][r] = 0.f; }

    const uint4* wb0 = reinterpret_cast<const uint4*>(Wb);

#pragma unroll 1
    for (int n = 0; n < 32; ++n) {
        int nbase = 0;
        asm volatile("" : "+v"(nbase));       // defeat LICM on the LDS A reads
        v8f acc0{}, acc1{};
        const uint4* wb = wb0 + ((size_t)(n * KSTEPS) * 32 + lane) * 2;
#pragma unroll
        for (int s = 0; s < KSTEPS; ++s) {
            uint4 bl = wb[0], bh = wb[1];
            wb += 64;                          // next K-step: 32 lanes * 32B
            uint4 al = *reinterpret_cast<const uint4*>(&A[nbase + arow * LSTR + s * 32 + aoff]);
            uint4 ah = *reinterpret_cast<const uint4*>(&A[nbase + arow * LSTR + s * 32 + aoff + 16]);
            v16bf bm = mkv(bl, bh);
            acc0 = __builtin_amdgcn_wmma_f32_16x16x32_bf16(
                false, a0[s], false, bm, (short)0, acc0, false, false);
            acc1 = __builtin_amdgcn_wmma_f32_16x16x32_bf16(
                false, mkv(al, ah), false, bm, (short)0, acc1, false, false);
        }
        const int   col = n * 16 + col0;
        const float vv  = Vv[col];
        const float* dr = dec_row + col;
#pragma unroll
        for (int r = 0; r < 8; ++r) {
            int gm0 = m0 + r + rbase;              // subtile 0 row
            float d0 = dr[(gm0 >= bnd) ? off1 : off0];
            eb[0][r] += vv * tanhf(acc0[r] + d0);
            int gm1 = gm0 + 16;                    // subtile 1 row
            float d1 = dr[(gm1 >= bnd) ? off1 : off0];
            eb[1][r] += vv * tanhf(acc1[r] + d1);
        }
    }
    // Reduce energy partials across the 16 lanes holding each row group
#pragma unroll
    for (int half = 0; half < 2; ++half)
#pragma unroll
        for (int r = 0; r < 8; ++r) {
            float e = eb[half][r];
            e += __shfl_xor(e, 1, 16);
            e += __shfl_xor(e, 2, 16);
            e += __shfl_xor(e, 4, 16);
            e += __shfl_xor(e, 8, 16);
            if (col0 == 0)
                energy[m0 + half * 16 + r + rbase] = e;
        }
}

// ---------------------------------------------------------------------------
// Kernel 5: masked (0/1 multiply, NOT -inf) softmax over all T, per batch
// ---------------------------------------------------------------------------
__global__ __launch_bounds__(256) void softmax_k(
    const float* __restrict__ energy, const int* __restrict__ xlens,
    float* __restrict__ aw)
{
    const int b = blockIdx.x, tid = threadIdx.x, L = xlens[b];
    __shared__ float red[8];
    float m = -3.0e38f;
    for (int t = tid; t < NT; t += 256) {
        float e = energy[b * NT + t];
        e = (t < L) ? e : 0.f;             // SHARPENING == 1.0
        m = fmaxf(m, e);
    }
#pragma unroll
    for (int o = 16; o >= 1; o >>= 1) m = fmaxf(m, __shfl_xor(m, o, 32));
    if ((tid & 31) == 0) red[tid >> 5] = m;
    __syncthreads();
    float M = red[0];
#pragma unroll
    for (int i = 1; i < 8; ++i) M = fmaxf(M, red[i]);
    __syncthreads();
    float s = 0.f;
    for (int t = tid; t < NT; t += 256) {
        float e = energy[b * NT + t];
        e = (t < L) ? e : 0.f;
        s += __expf(e - M);
    }
#pragma unroll
    for (int o = 16; o >= 1; o >>= 1) s += __shfl_xor(s, o, 32);
    if ((tid & 31) == 0) red[tid >> 5] = s;
    __syncthreads();
    float S = 0.f;
#pragma unroll
    for (int i = 0; i < 8; ++i) S += red[i];
    const float inv = 1.0f / S;
    for (int t = tid; t < NT; t += 256) {
        float e = energy[b * NT + t];
        e = (t < L) ? e : 0.f;
        aw[b * NT + t] = __expf(e - M) * inv;
    }
}

// ---------------------------------------------------------------------------
// Kernel 6: context[b,e] = sum_t aw[b,t] * enc[b,t,e]   (enc is L2-resident)
// ---------------------------------------------------------------------------
__global__ __launch_bounds__(128) void context_k(
    const float* __restrict__ enc, const float* __restrict__ aw,
    float* __restrict__ ctx)
{
    const int b = blockIdx.y;
    const int e = blockIdx.x * 128 + threadIdx.x;
    const float* ep = enc + (size_t)b * NT * NE + e;
    const float* ap = aw + b * NT;
    float acc = 0.f;
#pragma unroll 4
    for (int t = 0; t < NT; ++t) acc = fmaf(ap[t], ep[(size_t)t * NE], acc);
    ctx[b * NE + e] = acc;
}

// ---------------------------------------------------------------------------
extern "C" void kernel_launch(void* const* d_in, const int* in_sizes, int n_in,
                              void* d_out, int out_size, void* d_ws, size_t ws_size,
                              hipStream_t stream)
{
    const float* enc   = (const float*)d_in[0];   // [B,T,E]
    const float* dec   = (const float*)d_in[1];   // [B,1,D]
    const float* awst  = (const float*)d_in[2];   // [B,T,1]
    const int*   xlen  = (const int*)  d_in[3];   // [B]
    const float* Wenc  = (const float*)d_in[4];   // [E,A]
    const float* benc  = (const float*)d_in[5];   // [A]
    const float* Wdec  = (const float*)d_in[6];   // [D,A]
    const float* Wconv = (const float*)d_in[7];   // [C,A]
    const float* Vv    = (const float*)d_in[8];   // [A,1]
    const float* convw = (const float*)d_in[9];   // [C,1,K]

    float* ctx = (float*)d_out;                   // [B,1,E] -> 16384 floats
    float* awo = (float*)d_out + NB * NE;         // [B,T,1] -> 48000 floats

    char* w = (char*)d_ws;
    unsigned short* Wb   = (unsigned short*)w;  w += (size_t)KX * NA * 2;        // 557056 B
    unsigned short* cf16 = (unsigned short*)w;  w += (size_t)ROWS * 16 * 2;      // 1536000 B
    float* dec_row       = (float*)w;           w += (size_t)NB * NA * 4;        // 65536 B
    float* energy        = (float*)w;           w += (size_t)ROWS * 4;           // 192000 B
    (void)ws_size; (void)in_sizes; (void)n_in; (void)out_size;

    prep_wb   <<<1088, 256, 0, stream>>>(Wenc, Wconv, Wb);
    decrow_k  <<<dim3(2, NB), 256, 0, stream>>>(dec, Wdec, benc, dec_row);
    conv_loc  <<<dim3(6, NB), 256, 0, stream>>>(awst, convw, cf16);
    energy_gemm<<<ROWS / 64, 64, 0, stream>>>(enc, Wb, cf16, dec_row, Vv, energy);
    softmax_k <<<NB, 256, 0, stream>>>(energy, xlen, awo);
    context_k <<<dim3(NE / 128, NB), 128, 0, stream>>>(enc, awo, ctx);
}
